// RQVAE_21749714387653
// MI455X (gfx1250) — compile-verified, matrix operand order checked by reading, not run on hardware
//
#include <hip/hip_runtime.h>
#include <hip/hip_bf16.h>

// ---------------------------------------------------------------------------
// RQ-VAE forward for MI455X (gfx1250): f16 WMMA GEMMs (v_wmma_f32_16x16x32_f16)
// with double-buffered async global->LDS staging; fused L2-resident Sinkhorn
// (one R+W sweep of Q per iteration instead of two).
// ---------------------------------------------------------------------------

typedef _Float16 v16h __attribute__((ext_vector_type(16)));
typedef float    v8f  __attribute__((ext_vector_type(8)));

#define NB  131072   // batch rows
#define IND 768
#define H1D 512
#define H2D 256
#define ED  32
#define KD  256      // codebook entries
#define LVL 3
#define SKI 100

// ---------------- async global->LDS copy (CDNA5), guarded -------------------
#if __has_builtin(__builtin_amdgcn_global_load_async_to_lds_b128) && \
    __has_builtin(__builtin_amdgcn_s_wait_asynccnt)
#define USE_ASYNC_LDS 1
// Builtin prototype (from hipcc diagnostic): (int4 as1*, int4 as3*, Ii, Ii)
typedef int v4i __attribute__((vector_size(16)));
typedef __attribute__((address_space(1))) v4i* gv4_ptr;
typedef __attribute__((address_space(3))) v4i* lv4_ptr;
#else
#define USE_ASYNC_LDS 0
#endif

__device__ __forceinline__ void async_copy16(const _Float16* g, _Float16* l) {
#if USE_ASYNC_LDS
  // AS1 = full 64-bit global VA; AS3 = low 32 bits of the flat LDS address
  __builtin_amdgcn_global_load_async_to_lds_b128(
      (gv4_ptr)(uintptr_t)g, (lv4_ptr)(unsigned)(uintptr_t)l, 0, 0);
#else
  *reinterpret_cast<uint4*>(l) = *reinterpret_cast<const uint4*>(g);
#endif
}
__device__ __forceinline__ void async_wait() {
#if USE_ASYNC_LDS
  __builtin_amdgcn_s_wait_asynccnt(0);
#endif
}

union FragH { v16h v; unsigned u[8]; };

// 16x32 f16 WMMA operand fragment from a row-major LDS tile (rows = M for A,
// rows = N for the N-major-staged B). ISA 7.12.2: lane<16 holds K{0..7,16..23},
// lane>=16 holds K{8..15,24..31}.
__device__ __forceinline__ void load_frag(FragH& f, const _Float16* tile,
                                          int row, int ldk, int half_sel) {
  const unsigned* p = reinterpret_cast<const unsigned*>(tile + row * ldk);
#pragma unroll
  for (int vv = 0; vv < 8; ++vv) {
    int kb = ((vv & 4) ? 16 : 0) + half_sel * 8 + (vv & 3) * 2;
    f.u[vv] = p[kb >> 1];
  }
}

// Order-preserving float <-> uint mapping for atomic min/max on floats.
__device__ __forceinline__ unsigned f2ord(float f) {
  unsigned u = __float_as_uint(f);
  return (u & 0x80000000u) ? ~u : (u | 0x80000000u);
}
__device__ __forceinline__ float ord2f(unsigned u) {
  return (u & 0x80000000u) ? __uint_as_float(u ^ 0x80000000u)
                           : __uint_as_float(~u);
}

// ---------------------------------------------------------------------------
// WMMA GEMM:  C[M,Nd] = act( A[M,Kd] (f16) * Bt[Nd,Kd]^T (f16) + bias )
// Double-buffered LDS; async stage of tile k+1 overlaps WMMA on tile k.
// ---------------------------------------------------------------------------
template <int BM, int BN, int WR, int WC, bool RELU, bool OUT16>
__global__ __launch_bounds__(256) void gemm_wmma_kernel(
    const _Float16* __restrict__ A, const _Float16* __restrict__ Bt,
    const float* __restrict__ bias, _Float16* __restrict__ C16,
    float* __restrict__ C32, int M, int Kd, int Nd) {
  constexpr int LDK = 40;  // f16 row stride (80B rows, 16B aligned)
  constexpr int TM  = BM / (WR * 16);
  constexpr int TN  = BN / (WC * 16);
  __shared__ __align__(16) _Float16 As[2][BM * LDK];
  __shared__ __align__(16) _Float16 Bs[2][BN * LDK];

  const int    tid  = threadIdx.x;
  const size_t m0   = (size_t)blockIdx.x * BM;
  const size_t n0   = (size_t)blockIdx.y * BN;
  const int    wave = tid >> 5, lane = tid & 31;
  const int    wr   = wave % WR, wc = wave / WR;
  const int    hs   = lane >> 4, l15 = lane & 15;

  auto stage = [&](int buf, int k0) {
    for (int i = tid; i < BM * 4; i += 256) {
      int r = i >> 2, s = i & 3;
      async_copy16(&A[(m0 + r) * Kd + k0 + s * 8], &As[buf][r * LDK + s * 8]);
    }
    for (int i = tid; i < BN * 4; i += 256) {
      int r = i >> 2, s = i & 3;
      async_copy16(&Bt[(n0 + r) * Kd + k0 + s * 8], &Bs[buf][r * LDK + s * 8]);
    }
  };

  v8f acc[TM][TN];
#pragma unroll
  for (int a = 0; a < TM; ++a)
#pragma unroll
    for (int b = 0; b < TN; ++b)
#pragma unroll
      for (int j = 0; j < 8; ++j) acc[a][b][j] = 0.0f;

  stage(0, 0);
  async_wait();
  __syncthreads();

  const int nk = Kd >> 5;
  for (int ks = 0; ks < nk; ++ks) {
    const int cur = ks & 1;
    if (ks + 1 < nk) stage(cur ^ 1, (ks + 1) * 32);  // overlap with WMMA below

    const _Float16* Ac = As[cur];
    const _Float16* Bc = Bs[cur];
    FragH af[TM], bf[TN];
#pragma unroll
    for (int tm = 0; tm < TM; ++tm)
      load_frag(af[tm], Ac, (wr * TM + tm) * 16 + l15, LDK, hs);
#pragma unroll
    for (int tn = 0; tn < TN; ++tn)
      load_frag(bf[tn], Bc, (wc * TN + tn) * 16 + l15, LDK, hs);
#pragma unroll
    for (int tm = 0; tm < TM; ++tm)
#pragma unroll
      for (int tn = 0; tn < TN; ++tn)
        acc[tm][tn] = __builtin_amdgcn_wmma_f32_16x16x32_f16(
            false, af[tm].v, false, bf[tn].v, (short)0, acc[tm][tn],
            false, false);

    async_wait();     // our async writes into buf^1 have landed
    __syncthreads();  // everyone done reading buf / writing buf^1
  }

  // Epilogue per ISA C layout: VGPR r -> row r (lanes 0-15) / r+8 (lanes 16-31)
#pragma unroll
  for (int tm = 0; tm < TM; ++tm) {
#pragma unroll
    for (int tn = 0; tn < TN; ++tn) {
      size_t rbase = m0 + (size_t)(wr * TM + tm) * 16 + hs * 8;
      int    col   = (int)n0 + (wc * TN + tn) * 16 + l15;
      float  bv    = bias[col];
#pragma unroll
      for (int r = 0; r < 8; ++r) {
        float v = acc[tm][tn][r] + bv;
        if (RELU) v = fmaxf(v, 0.0f);
        size_t o = (rbase + r) * (size_t)Nd + col;
        if (OUT16) C16[o] = (_Float16)v;
        else       C32[o] = v;
      }
    }
  }
}

// ---------------------------------------------------------------------------
// VQ distance: d[n,k] = |r_n|^2 + |cb_k|^2 - 2 r_n.cb_k (K=E=32 -> one WMMA).
// Also reduces global min/max of d via ordered-uint atomics.
// ---------------------------------------------------------------------------
__global__ __launch_bounds__(256) void vq_dist_kernel(
    const float* __restrict__ resid, const float* __restrict__ cb,
    float* __restrict__ dmat, unsigned* __restrict__ mm) {
  constexpr int LDK = 40;
  __shared__ __align__(16) _Float16 As[128 * LDK];
  __shared__ __align__(16) _Float16 Cs[256 * LDK];
  __shared__ float rn2[128];
  __shared__ float cn2[256];
  const int    tid = threadIdx.x;
  const size_t m0  = (size_t)blockIdx.x * 128;

  {  // stage residual rows (fp32 -> f16) + fp32 row norms
    int r = tid >> 1, s = tid & 1;
    const float* src = resid + (m0 + r) * ED + s * 16;
    _Float16*    dst = As + r * LDK + s * 16;
    float ss = 0.f;
#pragma unroll
    for (int j = 0; j < 16; ++j) { float x = src[j]; ss += x * x; dst[j] = (_Float16)x; }
    ss += __shfl_xor(ss, 1);
    if (s == 0) rn2[r] = ss;
  }
  {  // stage codebook (fp32 -> f16) + fp32 code norms
    const float* src = cb + tid * ED;
    _Float16*    dst = Cs + tid * LDK;
    float ss = 0.f;
#pragma unroll
    for (int j = 0; j < ED; ++j) { float x = src[j]; ss += x * x; dst[j] = (_Float16)x; }
    cn2[tid] = ss;
  }
  __syncthreads();

  const int wave = tid >> 5, lane = tid & 31, hs = lane >> 4, l15 = lane & 15;
  FragH af;
  load_frag(af, As, wave * 16 + l15, LDK, hs);
  float lmin = 3.4e38f, lmax = -3.4e38f;
  for (int ct = 0; ct < 16; ++ct) {
    FragH bf;
    load_frag(bf, Cs, ct * 16 + l15, LDK, hs);
    v8f c;
#pragma unroll
    for (int j = 0; j < 8; ++j) c[j] = 0.f;
    c = __builtin_amdgcn_wmma_f32_16x16x32_f16(false, af.v, false, bf.v,
                                               (short)0, c, false, false);
    int   k  = ct * 16 + l15;
    float ck = cn2[k];
#pragma unroll
    for (int r = 0; r < 8; ++r) {
      int   m  = wave * 16 + hs * 8 + r;
      float dv = rn2[m] + ck - 2.0f * c[r];
      dmat[(m0 + m) * KD + k] = dv;
      lmin = fminf(lmin, dv);
      lmax = fmaxf(lmax, dv);
    }
  }
  atomicMin(&mm[0], f2ord(lmin));
  atomicMax(&mm[1], f2ord(lmax));
}

// ---------------------------------------------------------------------------
// Small helper kernels
// ---------------------------------------------------------------------------
__global__ void f32_to_f16_kernel(const float* __restrict__ in,
                                  _Float16* __restrict__ out, size_t n) {
  size_t i = (size_t)blockIdx.x * blockDim.x + threadIdx.x;
  size_t st = (size_t)gridDim.x * blockDim.x;
  for (; i < n; i += st) out[i] = (_Float16)in[i];
}

// W[Kd,Nd] fp32 -> Wt[Nd,Kd] f16 (N-major x K for B-operand staging)
__global__ void wtrans_kernel(const float* __restrict__ in,
                              _Float16* __restrict__ out, int Kd, int Nd) {
  size_t tot = (size_t)Kd * Nd;
  size_t i = (size_t)blockIdx.x * blockDim.x + threadIdx.x;
  size_t st = (size_t)gridDim.x * blockDim.x;
  for (; i < tot; i += st) {
    int k = (int)(i / Nd), n = (int)(i % Nd);
    out[(size_t)n * Kd + k] = (_Float16)in[i];
  }
}

__global__ void resid_init_kernel(const float* __restrict__ z,
                                  float* __restrict__ resid,
                                  float* __restrict__ zq, size_t n) {
  size_t i = (size_t)blockIdx.x * blockDim.x + threadIdx.x;
  size_t st = (size_t)gridDim.x * blockDim.x;
  for (; i < n; i += st) { resid[i] = z[i]; zq[i] = 0.0f; }
}

__global__ void scalar_init_kernel(float* loss) {
  if (threadIdx.x == 0 && blockIdx.x == 0) loss[0] = 0.0f;
}

__global__ void level_init_kernel(unsigned* mm, int* badflag) {
  if (threadIdx.x == 0 && blockIdx.x == 0) {
    mm[0] = 0xFFFFFFFFu;  // ordered +inf -> min identity
    mm[1] = 0u;           // ordered -inf -> max identity
    badflag[0] = 0;
  }
}

__global__ void sk_prep_kernel(const unsigned* __restrict__ mm,
                               float* __restrict__ params) {
  if (threadIdx.x == 0 && blockIdx.x == 0) {
    float mn  = ord2f(mm[0]);
    float mx  = ord2f(mm[1]);
    float mid = 0.5f * (mx + mn);
    float amp = mx - mid + 1e-5f;
    params[0] = mid;
    params[1] = -1.0f / (amp * 0.003f);  // Q = exp((d-mid)*coef)
  }
}

__global__ void sk_reset_kernel(float* __restrict__ colsum) {
  colsum[threadIdx.x] = 0.0f;
}

// colsum -> column scale for the next sweep; reset accumulator.
__global__ void sk_update_inv_kernel(float* __restrict__ invc,
                                     float* __restrict__ colsum) {
  int t = threadIdx.x;
  invc[t] = 1.0f / (colsum[t] * (float)KD);
  colsum[t] = 0.0f;
}

// Fused Sinkhorn sweep: apply previous column scale (or exp() init on the
// first sweep), row-normalize (x 1/(rowsum*N)), write back, accumulate the
// next column sums (LDS-combined; one atomicAdd per column per block).
// One R+W pass over Q per iteration. Block = 64 rows (8 waves x 8 rows).
template <bool FIRST>
__global__ __launch_bounds__(256) void sk_fused_kernel(
    const float* __restrict__ src, float* __restrict__ Q,
    const float* __restrict__ invc, const float* __restrict__ params,
    float* __restrict__ colsum) {
  __shared__ float sinv[256];
  __shared__ float scol[8 * 256];
  const int t = threadIdx.x;
  sinv[t] = FIRST ? 1.0f : invc[t];
  __syncthreads();
  float mid = 0.f, coef = 0.f;
  if (FIRST) { mid = params[0]; coef = params[1]; }

  const int wave = t >> 5, lane = t & 31;
  float ca[8];
#pragma unroll
  for (int j = 0; j < 8; ++j) ca[j] = 0.f;
  const size_t row0 = (size_t)blockIdx.x * 64 + wave * 8;

  for (int i = 0; i < 8; ++i) {
    const float* ps = src + (row0 + i) * KD + lane * 8;
    float4 a = *reinterpret_cast<const float4*>(ps);
    float4 b = *reinterpret_cast<const float4*>(ps + 4);
    float v[8] = {a.x, a.y, a.z, a.w, b.x, b.y, b.z, b.w};
    if (FIRST) {
#pragma unroll
      for (int j = 0; j < 8; ++j) v[j] = __expf((v[j] - mid) * coef);
    } else {
#pragma unroll
      for (int j = 0; j < 8; ++j) v[j] *= sinv[lane * 8 + j];
    }
    float s = 0.f;
#pragma unroll
    for (int j = 0; j < 8; ++j) s += v[j];
#pragma unroll
    for (int o = 16; o > 0; o >>= 1) s += __shfl_xor(s, o);
    float rinv = 1.0f / (s * (float)NB);
#pragma unroll
    for (int j = 0; j < 8; ++j) { v[j] *= rinv; ca[j] += v[j]; }
    float* pd = Q + (row0 + i) * KD + lane * 8;
    *reinterpret_cast<float4*>(pd)     = make_float4(v[0], v[1], v[2], v[3]);
    *reinterpret_cast<float4*>(pd + 4) = make_float4(v[4], v[5], v[6], v[7]);
  }
#pragma unroll
  for (int j = 0; j < 8; ++j) scol[wave * 256 + lane * 8 + j] = ca[j];
  __syncthreads();
  float s = 0.f;
#pragma unroll
  for (int w = 0; w < 8; ++w) s += scol[w * 256 + t];
  atomicAdd(&colsum[t], s);
}

// Finiteness check of (Q * pending column scale).
__global__ __launch_bounds__(256) void finite_check_kernel(
    const float* __restrict__ Q, const float* __restrict__ invc,
    int* __restrict__ flag, size_t n) {
  __shared__ float sinv[256];
  sinv[threadIdx.x] = invc[threadIdx.x];
  __syncthreads();
  size_t i = (size_t)blockIdx.x * blockDim.x + threadIdx.x;
  size_t st = (size_t)gridDim.x * blockDim.x;
  int bad = 0;
  for (; i < n; i += st) {
    float v = Q[i] * sinv[i & 255];
    if (!(fabsf(v) <= 3.0e38f)) bad = 1;  // catches NaN and Inf
  }
  if (bad) atomicOr(flag, 1);
}

// Per-row assignment (argmax Q*invc, or argmin d on the bad path), codebook
// gather, residual/zq update, loss accumulation. One wave per row (E=32).
__global__ __launch_bounds__(256) void vq_assign_kernel(
    const float* __restrict__ Q, const float* __restrict__ invc,
    const float* __restrict__ dmat, const float* __restrict__ cb,
    float* __restrict__ resid, float* __restrict__ zq, int* __restrict__ idxbuf,
    const int* __restrict__ badflag, float* __restrict__ loss_acc, int level) {
  __shared__ float sinv[256];
  sinv[threadIdx.x] = invc[threadIdx.x];
  __syncthreads();
  const int wave = threadIdx.x >> 5, lane = threadIdx.x & 31;
  size_t row = (size_t)blockIdx.x * 8 + wave;
  bool  bad = (*badflag) != 0;
  const float* p = (bad ? dmat : Q) + row * KD + lane * 8;
  float bestv = -3.4e38f;
  int   bestk = 0;
#pragma unroll
  for (int j = 0; j < 8; ++j) {
    int   k = lane * 8 + j;
    float v = bad ? -p[j] : p[j] * sinv[k];  // argmin d == argmax(-d)
    if (v > bestv) { bestv = v; bestk = k; }  // strict > keeps first max
  }
#pragma unroll
  for (int o = 16; o > 0; o >>= 1) {
    float ov = __shfl_xor(bestv, o);
    int   ok = __shfl_xor(bestk, o);
    if (ov > bestv || (ov == bestv && ok < bestk)) { bestv = ov; bestk = ok; }
  }
  float  c    = cb[(size_t)bestk * ED + lane];
  size_t off  = row * ED + lane;
  float  rold = resid[off];
  float  diff = c - rold;
  resid[off] = rold - c;
  zq[off] += c;
  float ps = diff * diff;
#pragma unroll
  for (int o = 16; o > 0; o >>= 1) ps += __shfl_xor(ps, o);
  if (lane == 0) {
    idxbuf[row * LVL + level] = bestk;
    atomicAdd(loss_acc, ps);
  }
}

// Pack scalar loss + indices behind the decoder output.
__global__ void pack_kernel(float* __restrict__ tail,
                            const float* __restrict__ loss_acc,
                            const int* __restrict__ idxbuf, size_t nidx) {
  size_t i = (size_t)blockIdx.x * blockDim.x + threadIdx.x;
  size_t st = (size_t)gridDim.x * blockDim.x;
  for (; i < nidx + 1; i += st) {
    if (i == 0)
      tail[0] = loss_acc[0] * 1.25f / (float)((size_t)NB * ED * LVL);
    else
      tail[i] = (float)idxbuf[i - 1];
  }
}

// ---------------------------------------------------------------------------
// Host launch
// ---------------------------------------------------------------------------
extern "C" void kernel_launch(void* const* d_in, const int* in_sizes, int n_in,
                              void* d_out, int out_size, void* d_ws, size_t ws_size,
                              hipStream_t stream) {
  const float* x   = (const float*)d_in[0];
  const float* ew0 = (const float*)d_in[1];  const float* eb0 = (const float*)d_in[2];
  const float* ew1 = (const float*)d_in[3];  const float* eb1 = (const float*)d_in[4];
  const float* ew2 = (const float*)d_in[5];  const float* eb2 = (const float*)d_in[6];
  const float* dw0 = (const float*)d_in[7];  const float* db0 = (const float*)d_in[8];
  const float* dw1 = (const float*)d_in[9];  const float* db1 = (const float*)d_in[10];
  const float* dw2 = (const float*)d_in[11]; const float* db2 = (const float*)d_in[12];
  const float* cbs = (const float*)d_in[13];
  float* out = (float*)d_out;

  const size_t N = NB;
  char*  W   = (char*)d_ws;
  size_t off = 0;
  auto take = [&](size_t bytes) {
    size_t r = off;
    off += (bytes + 255) & ~(size_t)255;
    return r;
  };
  _Float16* xh   = (_Float16*)(W + take(N * IND * 2));  // dead after enc L0 -> Q
  _Float16* h1h  = (_Float16*)(W + take(N * H1D * 2));  // aliased: dmat, dec g2
  _Float16* h2h  = (_Float16*)(W + take(N * H2D * 2));  // aliased: dec g1
  float*    z    = (float*)(W + take(N * ED * 4));
  float*    resid= (float*)(W + take(N * ED * 4));
  float*    zq   = (float*)(W + take(N * ED * 4));
  _Float16* zqh  = (_Float16*)(W + take(N * ED * 2));
  _Float16* ew0t = (_Float16*)(W + take((size_t)IND * H1D * 2));
  _Float16* ew1t = (_Float16*)(W + take((size_t)H1D * H2D * 2));
  _Float16* ew2t = (_Float16*)(W + take((size_t)H2D * ED * 2));
  _Float16* dw0t = (_Float16*)(W + take((size_t)ED * H2D * 2));
  _Float16* dw1t = (_Float16*)(W + take((size_t)H2D * H1D * 2));
  _Float16* dw2t = (_Float16*)(W + take((size_t)H1D * IND * 2));
  int*      idxb = (int*)(W + take(N * LVL * 4));
  float*    colsum = (float*)(W + take(256 * 4));
  float*    invc   = (float*)(W + take(256 * 4));
  unsigned* mm     = (unsigned*)(W + take(2 * 4));
  float*    params = (float*)(W + take(2 * 4));
  int*      badflag= (int*)(W + take(4));
  float*    lossac = (float*)(W + take(4));
  // Aliases (disjoint lifetimes): Q lives in xh's region, dmat in h1h's.
  float* Q    = (float*)xh;    // N*KD*4 = 134MB <= N*IND*2 = 201MB
  float* dmat = (float*)h1h;   // N*KD*4 == N*H1D*2 exactly
  _Float16* g1h = h2h;
  _Float16* g2h = h1h;

  const size_t nQ = N * KD;

  scalar_init_kernel<<<1, 64, 0, stream>>>(lossac);

  // ---- precision conversion ----
  f32_to_f16_kernel<<<2048, 256, 0, stream>>>(x, xh, N * IND);
  wtrans_kernel<<<512, 256, 0, stream>>>(ew0, ew0t, IND, H1D);
  wtrans_kernel<<<512, 256, 0, stream>>>(ew1, ew1t, H1D, H2D);
  wtrans_kernel<<<64,  256, 0, stream>>>(ew2, ew2t, H2D, ED);
  wtrans_kernel<<<64,  256, 0, stream>>>(dw0, dw0t, ED, H2D);
  wtrans_kernel<<<512, 256, 0, stream>>>(dw1, dw1t, H2D, H1D);
  wtrans_kernel<<<512, 256, 0, stream>>>(dw2, dw2t, H1D, IND);

  // ---- encoder ----
  gemm_wmma_kernel<128, 64, 4, 2, true, true>
      <<<dim3(N / 128, H1D / 64), 256, 0, stream>>>(xh, ew0t, eb0, h1h, nullptr,
                                                    (int)N, IND, H1D);
  gemm_wmma_kernel<128, 64, 4, 2, true, true>
      <<<dim3(N / 128, H2D / 64), 256, 0, stream>>>(h1h, ew1t, eb1, h2h, nullptr,
                                                    (int)N, H1D, H2D);
  gemm_wmma_kernel<128, 32, 8, 1, false, false>
      <<<dim3(N / 128, 1), 256, 0, stream>>>(h2h, ew2t, eb2, nullptr, z,
                                             (int)N, H2D, ED);

  resid_init_kernel<<<1024, 256, 0, stream>>>(z, resid, zq, N * ED);

  // ---- residual VQ levels ----
  for (int l = 0; l < LVL; ++l) {
    const float* cb = cbs + (size_t)l * KD * ED;
    level_init_kernel<<<1, 64, 0, stream>>>(mm, badflag);
    vq_dist_kernel<<<N / 128, 256, 0, stream>>>(resid, cb, dmat, mm);
    sk_prep_kernel<<<1, 64, 0, stream>>>(mm, params);
    sk_reset_kernel<<<1, 256, 0, stream>>>(colsum);
    // Sweep 0: exp-init + row-normalize (global sum cancels in row norm).
    sk_fused_kernel<true><<<N / 64, 256, 0, stream>>>(dmat, Q, invc, params,
                                                      colsum);
    sk_update_inv_kernel<<<1, 256, 0, stream>>>(invc, colsum);
    // Sweeps 1..99: fused col-scale(prev) + row-normalize.
    for (int it = 1; it < SKI; ++it) {
      sk_fused_kernel<false><<<N / 64, 256, 0, stream>>>(Q, Q, invc, params,
                                                         colsum);
      sk_update_inv_kernel<<<1, 256, 0, stream>>>(invc, colsum);
    }
    // Final column scale (invc) is applied lazily below.
    finite_check_kernel<<<2048, 256, 0, stream>>>(Q, invc, badflag, nQ);
    vq_assign_kernel<<<N / 8, 256, 0, stream>>>(Q, invc, dmat, cb, resid, zq,
                                                idxb, badflag, lossac, l);
  }

  // ---- decoder ----
  f32_to_f16_kernel<<<512, 256, 0, stream>>>(zq, zqh, N * ED);
  gemm_wmma_kernel<128, 64, 4, 2, true, true>
      <<<dim3(N / 128, H2D / 64), 256, 0, stream>>>(zqh, dw0t, db0, g1h, nullptr,
                                                    (int)N, ED, H2D);
  gemm_wmma_kernel<128, 64, 4, 2, true, true>
      <<<dim3(N / 128, H1D / 64), 256, 0, stream>>>(g1h, dw1t, db1, g2h, nullptr,
                                                    (int)N, H2D, H1D);
  gemm_wmma_kernel<128, 64, 4, 2, false, false>
      <<<dim3(N / 128, IND / 64), 256, 0, stream>>>(g2h, dw2t, db2, nullptr, out,
                                                    (int)N, H1D, IND);

  // ---- pack loss + indices ----
  pack_kernel<<<1024, 256, 0, stream>>>(out + N * IND, lossac, idxb, N * LVL);

  (void)in_sizes; (void)n_in; (void)out_size; (void)ws_size;
}
